// TransformerBlock_64828236366450
// MI455X (gfx1250) — matrix operand order repaired and checked by reading, compiled, not run on hardware
//
#include <hip/hip_runtime.h>
#include <stdint.h>

#define NNODES 50000
#define MEDGES 800000
#define DIM    256
#define HEADS  8
#define BM     128
#define BN     64
#define BK     32
#define NPAD   50048   // 391 * 128, padded row count for GEMM A operands

typedef __attribute__((ext_vector_type(16))) __bf16 v16bf;
typedef __attribute__((ext_vector_type(8)))  float  v8f;
typedef __attribute__((ext_vector_type(4)))  int    v4i;

// ---- async LDS copy path (CDNA5), guarded so the file always compiles ----
#if defined(__has_builtin)
#if __has_builtin(__builtin_amdgcn_global_load_async_to_lds_b128) && \
    __has_builtin(__builtin_amdgcn_s_wait_asynccnt)
#define HAVE_ASYNC_LDS 1
#endif
#endif
#ifndef HAVE_ASYNC_LDS
#define HAVE_ASYNC_LDS 0
#endif

#if HAVE_ASYNC_LDS
typedef __attribute__((address_space(1))) v4i* g4_ptr;   // global int4*
typedef __attribute__((address_space(3))) v4i* l4_ptr;   // LDS int4*
__device__ __forceinline__ void async_copy16(const void* g, void* l) {
    __builtin_amdgcn_global_load_async_to_lds_b128((g4_ptr)(void*)g, (l4_ptr)l, 0, 0);
}
#endif

__device__ __forceinline__ float bf2f(unsigned short b) {
    unsigned u = ((unsigned)b) << 16;
    return __builtin_bit_cast(float, u);
}
__device__ __forceinline__ unsigned short f2bf(float f) {
    unsigned u = __builtin_bit_cast(unsigned, f);
    unsigned r = u + 0x7FFFu + ((u >> 16) & 1u);   // round-to-nearest-even
    return (unsigned short)(r >> 16);
}

// ---------------------------------------------------------------------------
// One-shot conversion kernels (hoist all fp32->bf16 work out of the GEMMs)
// ---------------------------------------------------------------------------
__global__ __launch_bounds__(256)
void conv_bf16_k(const float* __restrict__ src, unsigned short* __restrict__ dst,
                 int n8, int relu)   // n8 = #elements / 8
{
    int gid = blockIdx.x * 256 + threadIdx.x;
    if (gid < n8) {
        const float4* s4 = (const float4*)(src + (size_t)gid * 8);
        float4 a = s4[0], b = s4[1];
        if (relu) {
            a.x = fmaxf(a.x, 0.f); a.y = fmaxf(a.y, 0.f);
            a.z = fmaxf(a.z, 0.f); a.w = fmaxf(a.w, 0.f);
            b.x = fmaxf(b.x, 0.f); b.y = fmaxf(b.y, 0.f);
            b.z = fmaxf(b.z, 0.f); b.w = fmaxf(b.w, 0.f);
        }
        union { unsigned short h[8]; uint4 v; } o;
        o.h[0] = f2bf(a.x); o.h[1] = f2bf(a.y); o.h[2] = f2bf(a.z); o.h[3] = f2bf(a.w);
        o.h[4] = f2bf(b.x); o.h[5] = f2bf(b.y); o.h[6] = f2bf(b.z); o.h[7] = f2bf(b.w);
        ((uint4*)dst)[gid] = o.v;
    }
}

// transpose-convert one 256x256 weight matrix: Wt[n][k] = bf16(W[k][n])
__global__ __launch_bounds__(256)
void wt_conv_k(const float* __restrict__ W, unsigned short* __restrict__ Wt)
{
    int idx = blockIdx.x * 256 + threadIdx.x;   // 65536 total
    int nn = idx >> 8, kk = idx & 255;
    Wt[idx] = f2bf(W[(size_t)kk * DIM + nn]);
}

// ---------------------------------------------------------------------------
// WMMA GEMM:  C[n,256] = epilogue( A[n,256](bf16) @ Wt^T(bf16) + bias )
// A padded to >= gridDim.x*BM rows; Wt is n-major [256][256].
// 256 threads = 8 waves; each wave owns a 32x32 C tile (2x2 WMMA frags).
// ---------------------------------------------------------------------------
template<int OUT_BF16, int RELU_OUT, int ADD_RESID>
__global__ __launch_bounds__(256)
void gemm_wmma_k(const unsigned short* __restrict__ Ah,
                 const unsigned short* __restrict__ Wt,
                 const float* __restrict__ bias, void* __restrict__ Cv,
                 const float* __restrict__ resid, int Nrows)
{
    // 80-byte rows: 16B-aligned segments, conflict-free fragment reads
    __shared__ __align__(16) unsigned short As[BM][40];
    __shared__ __align__(16) unsigned short Bs[BN][40];

    const int tid  = threadIdx.x;
    const int lane = tid & 31, wave = tid >> 5;
    const int wm = wave >> 1, wn = wave & 1;     // 4x2 wave grid
    const int hh = lane >> 4, lr = lane & 15;
    const int blockRow = blockIdx.x * BM;
    const int colBase  = blockIdx.y * BN;

    v8f c[2][2] = {};

    for (int k0 = 0; k0 < DIM; k0 += BK) {
        __syncthreads();
        // ---- stage tiles: pure contiguous 16B copies, no guards ----
#if HAVE_ASYNC_LDS
        #pragma unroll
        for (int i = 0; i < 2; ++i) {
            int cidx = i * 256 + tid;            // 512 chunks for A (128 x 64B)
            int r = cidx >> 2, seg = cidx & 3;
            async_copy16(&Ah[(size_t)(blockRow + r) * DIM + k0 + seg * 8],
                         &As[r][seg * 8]);
        }
        {
            int r = tid >> 2, seg = tid & 3;     // 256 chunks for B (64 x 64B)
            async_copy16(&Wt[(size_t)(colBase + r) * DIM + k0 + seg * 8],
                         &Bs[r][seg * 8]);
        }
        __builtin_amdgcn_s_wait_asynccnt(0);
#else
        #pragma unroll
        for (int i = 0; i < 2; ++i) {
            int cidx = i * 256 + tid;
            int r = cidx >> 2, seg = cidx & 3;
            *(uint4*)&As[r][seg * 8] =
                *(const uint4*)&Ah[(size_t)(blockRow + r) * DIM + k0 + seg * 8];
        }
        {
            int r = tid >> 2, seg = tid & 3;
            *(uint4*)&Bs[r][seg * 8] =
                *(const uint4*)&Wt[(size_t)(colBase + r) * DIM + k0 + seg * 8];
        }
        if (k0 + BK < DIM)
            __builtin_prefetch(&Ah[(size_t)(blockRow + (tid >> 1)) * DIM + k0 + BK], 0, 1);
#endif
        __syncthreads();

        // ---- build fragments per ISA VGPR layouts ----
        union UA { v16bf v; unsigned u[8]; };
        UA a[2], b[2];
        #pragma unroll
        for (int mt = 0; mt < 2; ++mt) {
            int R = wm * 32 + mt * 16 + lr;
            #pragma unroll
            for (int vv = 0; vv < 8; ++vv) {
                // 16-bit A 16x32: vgpr v holds a K pair; lanes16-31 K +8
                int kb = (vv < 4 ? 2 * vv : 16 + 2 * (vv - 4)) + 8 * hh;
                a[mt].u[vv] = *(const unsigned*)&As[R][kb];
            }
        }
        #pragma unroll
        for (int nt = 0; nt < 2; ++nt) {
            int Cn = wn * 32 + nt * 16 + lr;
            #pragma unroll
            for (int j = 0; j < 8; ++j) {
                // 16-bit B 32x16: same K order as A (vgpr j holds a K pair; lanes16-31 K +8)
                b[nt].u[j] = *(const unsigned*)&Bs[Cn][(j < 4 ? 2 * j : 16 + 2 * (j - 4)) + 8 * hh];
            }
        }
        #pragma unroll
        for (int mt = 0; mt < 2; ++mt)
            #pragma unroll
            for (int nt = 0; nt < 2; ++nt)
                c[mt][nt] = __builtin_amdgcn_wmma_f32_16x16x32_bf16(
                    false, a[mt].v, false, b[nt].v, (short)0, c[mt][nt],
                    false, false);
    }

    // ---- epilogue: bias, ReLU, residual, store (row-guarded) ----
    #pragma unroll
    for (int mt = 0; mt < 2; ++mt) {
        #pragma unroll
        for (int nt = 0; nt < 2; ++nt) {
            int col  = colBase + wn * 32 + nt * 16 + lr;
            float bc = bias[col];
            #pragma unroll
            for (int r = 0; r < 8; ++r) {
                int grow = blockRow + wm * 32 + mt * 16 + 8 * hh + r;
                if (grow < Nrows) {
                    float v = c[mt][nt][r] + bc;
                    if (RELU_OUT)  v = v > 0.f ? v : 0.f;
                    if (ADD_RESID) v += resid[(size_t)grow * DIM + col];
                    if (OUT_BF16)
                        ((unsigned short*)Cv)[(size_t)grow * DIM + col] = f2bf(v);
                    else
                        ((float*)Cv)[(size_t)grow * DIM + col] = v;
                }
            }
        }
    }
}

// ---------------------------------------------------------------------------
// Edge phase kernels (one thread per (edge, head))
// ---------------------------------------------------------------------------
__global__ __launch_bounds__(256)
void edge_logits_k(const int* __restrict__ ei,
                   const unsigned short* __restrict__ Qb,
                   const unsigned short* __restrict__ Kb,
                   float* __restrict__ att, float* __restrict__ bmax)
{
    __shared__ float red[256];
    int gid = blockIdx.x * 256 + threadIdx.x;
    float mv = -3.4e38f;
    if (gid < MEDGES * HEADS) {
        int e = gid >> 3, h = gid & 7;
        int recv = ei[e], send = ei[MEDGES + e];
        const unsigned* qu = (const unsigned*)(Qb + (size_t)recv * DIM + h * 32);
        const unsigned* ku = (const unsigned*)(Kb + (size_t)send * DIM + h * 32);
        float s = 0.f;
        #pragma unroll
        for (int j = 0; j < 16; ++j) {
            unsigned qa = qu[j], ka = ku[j];
            s += bf2f((unsigned short)qa)         * bf2f((unsigned short)ka);
            s += bf2f((unsigned short)(qa >> 16)) * bf2f((unsigned short)(ka >> 16));
        }
        att[gid] = s;
        mv = s;
    }
    red[threadIdx.x] = mv;
    __syncthreads();
    for (int off = 128; off > 0; off >>= 1) {
        if (threadIdx.x < off)
            red[threadIdx.x] = fmaxf(red[threadIdx.x], red[threadIdx.x + off]);
        __syncthreads();
    }
    if (threadIdx.x == 0) bmax[blockIdx.x] = red[0];
}

__global__ __launch_bounds__(256)
void max_reduce_k(const float* __restrict__ bmax, int n, float* __restrict__ gmax)
{
    __shared__ float red[256];
    float m = -3.4e38f;
    for (int i = threadIdx.x; i < n; i += 256) m = fmaxf(m, bmax[i]);
    red[threadIdx.x] = m;
    __syncthreads();
    for (int off = 128; off > 0; off >>= 1) {
        if (threadIdx.x < off)
            red[threadIdx.x] = fmaxf(red[threadIdx.x], red[threadIdx.x + off]);
        __syncthreads();
    }
    if (threadIdx.x == 0) gmax[0] = red[0];
}

__global__ __launch_bounds__(256)
void edge_exp_k(const int* __restrict__ ei, float* __restrict__ att,
                const float* __restrict__ gmax, float* __restrict__ sums)
{
    int gid = blockIdx.x * 256 + threadIdx.x;
    if (gid < MEDGES * HEADS) {
        int e = gid >> 3, h = gid & 7;
        float a = __expf(att[gid] * 3.f / gmax[0]);
        att[gid] = a;
        atomicAdd(&sums[(size_t)ei[e] * HEADS + h], a);
    }
}

__global__ __launch_bounds__(256)
void edge_scatter_k(const int* __restrict__ ei, const float* __restrict__ att,
                    const float* __restrict__ sums, const float* __restrict__ Vb,
                    float* __restrict__ agg)
{
    int gid = blockIdx.x * 256 + threadIdx.x;
    if (gid < MEDGES * HEADS) {
        int e = gid >> 3, h = gid & 7;
        int recv = ei[e], send = ei[MEDGES + e];
        float w = att[gid] / sums[(size_t)recv * HEADS + h] * 0.17677669529663687f;
        const float4* v4 = (const float4*)(Vb + (size_t)send * DIM + h * 32);
        float* dst = agg + (size_t)recv * DIM + h * 32;
        #pragma unroll
        for (int j = 0; j < 8; ++j) {
            float4 vv = v4[j];
            atomicAdd(dst + 4 * j + 0, w * vv.x);
            atomicAdd(dst + 4 * j + 1, w * vv.y);
            atomicAdd(dst + 4 * j + 2, w * vv.z);
            atomicAdd(dst + 4 * j + 3, w * vv.w);
        }
    }
}

// ---------------------------------------------------------------------------
extern "C" void kernel_launch(void* const* d_in, const int* in_sizes, int n_in,
                              void* d_out, int out_size, void* d_ws, size_t ws_size,
                              hipStream_t stream)
{
    (void)in_sizes; (void)n_in; (void)out_size; (void)ws_size;

    const float* x    = (const float*)d_in[0];
    const int*   ei   = (const int*)  d_in[1];
    const float* Wk   = (const float*)d_in[2];
    const float* bk   = (const float*)d_in[3];
    const float* Wq   = (const float*)d_in[4];
    const float* bq   = (const float*)d_in[5];
    const float* Wv   = (const float*)d_in[6];
    const float* bv   = (const float*)d_in[7];
    const float* Wagg = (const float*)d_in[8];
    const float* bagg = (const float*)d_in[9];
    const float* Wff  = (const float*)d_in[10];
    const float* bff  = (const float*)d_in[11];
    float* out = (float*)d_out;

    char* p = (char*)d_ws;
    auto carve = [&](size_t bytes) -> char* {
        char* q = p; p += (bytes + 255) & ~(size_t)255; return q;
    };
    unsigned short* xh   = (unsigned short*)carve((size_t)NPAD * DIM * 2);
    unsigned short* WtQ  = (unsigned short*)carve((size_t)DIM * DIM * 2);
    unsigned short* WtK  = (unsigned short*)carve((size_t)DIM * DIM * 2);
    unsigned short* WtV  = (unsigned short*)carve((size_t)DIM * DIM * 2);
    unsigned short* WtA  = (unsigned short*)carve((size_t)DIM * DIM * 2);
    unsigned short* WtF  = (unsigned short*)carve((size_t)DIM * DIM * 2);
    unsigned short* Qb   = (unsigned short*)carve((size_t)NNODES * DIM * 2);
    unsigned short* Kb   = (unsigned short*)carve((size_t)NNODES * DIM * 2);
    float*          Vb   = (float*)         carve((size_t)NNODES * DIM * 4);
    float*          att  = (float*)         carve((size_t)MEDGES * HEADS * 4);
    float*          sums = (float*)         carve((size_t)NNODES * HEADS * 4);
    float*          agg  = (float*)         carve((size_t)NNODES * DIM * 4);
    unsigned short* aggH = (unsigned short*)carve((size_t)NPAD * DIM * 2);
    unsigned short* h1   = (unsigned short*)carve((size_t)NPAD * DIM * 2);
    float*          bmax = (float*)         carve((size_t)(MEDGES * HEADS / 256) * 4);
    float*          gmax = (float*)         carve(256);

    (void)hipMemsetAsync(sums, 0, (size_t)NNODES * HEADS * 4, stream);
    (void)hipMemsetAsync(agg,  0, (size_t)NNODES * DIM * 4, stream);

    dim3 blk(256);
    const int n8    = NNODES * DIM / 8;                 // 1.6M
    const int cGrid = (n8 + 255) / 256;

    // one-shot conversions
    conv_bf16_k<<<cGrid, blk, 0, stream>>>(x, xh, n8, 0);
    wt_conv_k<<<256, blk, 0, stream>>>(Wq,   WtQ);
    wt_conv_k<<<256, blk, 0, stream>>>(Wk,   WtK);
    wt_conv_k<<<256, blk, 0, stream>>>(Wv,   WtV);
    wt_conv_k<<<256, blk, 0, stream>>>(Wagg, WtA);
    wt_conv_k<<<256, blk, 0, stream>>>(Wff,  WtF);

    dim3 gGrid(NPAD / BM, DIM / BN);                    // 391 x 4

    gemm_wmma_k<1, 0, 0><<<gGrid, blk, 0, stream>>>(xh, WtQ, bq, Qb, nullptr, NNODES);
    gemm_wmma_k<1, 0, 0><<<gGrid, blk, 0, stream>>>(xh, WtK, bk, Kb, nullptr, NNODES);
    gemm_wmma_k<0, 0, 0><<<gGrid, blk, 0, stream>>>(xh, WtV, bv, Vb, nullptr, NNODES);

    const int eBlocks = MEDGES * HEADS / 256;           // 25000
    edge_logits_k <<<eBlocks, blk, 0, stream>>>(ei, Qb, Kb, att, bmax);
    max_reduce_k  <<<1,       blk, 0, stream>>>(bmax, eBlocks, gmax);
    edge_exp_k    <<<eBlocks, blk, 0, stream>>>(ei, att, gmax, sums);
    edge_scatter_k<<<eBlocks, blk, 0, stream>>>(ei, att, sums, Vb, agg);

    // aggH = bf16(relu(agg)); h1 = relu(aggH @ Wagg + bagg)
    conv_bf16_k<<<cGrid, blk, 0, stream>>>(agg, aggH, n8, 1);
    gemm_wmma_k<1, 1, 0><<<gGrid, blk, 0, stream>>>(aggH, WtA, bagg, h1, nullptr, NNODES);
    // out = x + relu(h1 @ Wff + bff)
    gemm_wmma_k<0, 1, 1><<<gGrid, blk, 0, stream>>>(h1, WtF, bff, out, x, NNODES);
}
